// QuantumTransformer_17171279250110
// MI455X (gfx1250) — matrix operand order repaired and checked by reading, not compiled, not run on hardware
//
#include <hip/hip_runtime.h>

// ---------------------------------------------------------------------------
// Types for CDNA5 WMMA (wave32): v_wmma_f32_16x16x32_bf16
// ---------------------------------------------------------------------------
typedef __attribute__((ext_vector_type(16))) __bf16          v16bf;
typedef __attribute__((ext_vector_type(16))) unsigned short  v16u;
typedef __attribute__((ext_vector_type(8)))  float           v8f;

union Frag { v16u v; unsigned u[8]; unsigned short s16[16]; };

static __device__ __forceinline__ unsigned short f2bf(float f) {
  unsigned u = __builtin_bit_cast(unsigned, f);
  u += 0x7FFFu + ((u >> 16) & 1u);               // round-to-nearest-even
  return (unsigned short)(u >> 16);
}

static __device__ __forceinline__ v8f wmma_bf16(v16u a, v16u b, v8f c) {
  // 8 args: (neg_a, A, neg_b, B, c_mod, C, reuse_a, reuse_b)
  return __builtin_amdgcn_wmma_f32_16x16x32_bf16(
      false, __builtin_bit_cast(v16bf, a),
      false, __builtin_bit_cast(v16bf, b),
      (short)0, c, false, false);
}

// Problem constants
#define MTOK 16384   // B*S
#define DMODEL 256
#define TD 768       // 3*D
#define FFD 2048
#define SEQ 1024

// ---------------------------------------------------------------------------
// f32 -> bf16 weight conversion
// ---------------------------------------------------------------------------
__global__ void cvt_f32_bf16(const float* __restrict__ src,
                             unsigned short* __restrict__ dst, int n) {
  int i = blockIdx.x * 256 + threadIdx.x;
  if (i < n) dst[i] = f2bf(src[i]);
}

// ---------------------------------------------------------------------------
// Embedding: h = (cos(theta)*rowsum(proj_w) + proj_b)*sqrt(D) + pos_enc
// ---------------------------------------------------------------------------
__global__ void embed_kernel(const int* __restrict__ x,
                             const float* __restrict__ proj_w,
                             const float* __restrict__ proj_b,
                             const float* __restrict__ pos,
                             float* __restrict__ h) {
  int idx = blockIdx.x * 256 + threadIdx.x;      // 0 .. MTOK*DMODEL-1
  int row = idx >> 8;
  int d   = idx & 255;
  int s   = row & (SEQ - 1);
  float theta = (float)x[row] * 0.15707963267948966f;   // pi/20
  float c = __cosf(theta);
  const float* pw = proj_w + d * 4;
  float pws = pw[0] + pw[1] + pw[2] + pw[3];
  h[idx] = (c * pws + proj_b[d]) * 16.0f + pos[s * DMODEL + d];
}

// ---------------------------------------------------------------------------
// Generic WMMA GEMM: Out[M,N] = act(A[M,K] @ W[N,K]^T + bias[N])
// Block: 256 threads = 8 waves (2 M-waves x 4 N-waves), block tile 64x256.
// Wave tile 32x64 -> 8 WMMAs per K-step from 2 A-frags + 4 B-frags.
// ---------------------------------------------------------------------------
template<bool AF32, bool OF32, bool RELU>
__global__ __launch_bounds__(256)
void gemm_wmma(const void* __restrict__ Aptr,
               const unsigned short* __restrict__ W,
               const float* __restrict__ bias,
               void* __restrict__ Out, int M, int N, int K) {
  __shared__ unsigned short sA[64 * 34];     // padded stride 34
  __shared__ unsigned short sB[256 * 34];

  const int tid  = threadIdx.x;
  const int lane = tid & 31;
  const int wave = tid >> 5;
  const int waveM = wave & 1, waveN = wave >> 1;         // 2 x 4 wave grid
  const int half = lane >> 4, l16 = lane & 15;
  const int blockRow = blockIdx.y * 64;
  const int blockCol = blockIdx.x * 256;

  const int rowA = tid >> 2, kcA = (tid & 3) * 8;        // A stage: 8 elems/thread

  const v8f vzero = {0.f,0.f,0.f,0.f,0.f,0.f,0.f,0.f};
  v8f acc[2][4];
  #pragma unroll
  for (int i = 0; i < 2; ++i)
    #pragma unroll
    for (int j = 0; j < 4; ++j) acc[i][j] = vzero;

  for (int k0 = 0; k0 < K; k0 += 32) {
    __syncthreads();
    { // stage A tile (64x32) as bf16
      unsigned* dst = (unsigned*)&sA[rowA * 34 + kcA];
      if constexpr (AF32) {
        const float* Af = (const float*)Aptr + (size_t)(blockRow + rowA) * K + (k0 + kcA);
        float4 f0 = ((const float4*)Af)[0];
        float4 f1 = ((const float4*)Af)[1];
        dst[0] = (unsigned)f2bf(f0.x) | ((unsigned)f2bf(f0.y) << 16);
        dst[1] = (unsigned)f2bf(f0.z) | ((unsigned)f2bf(f0.w) << 16);
        dst[2] = (unsigned)f2bf(f1.x) | ((unsigned)f2bf(f1.y) << 16);
        dst[3] = (unsigned)f2bf(f1.z) | ((unsigned)f2bf(f1.w) << 16);
        if (k0 + 32 < K) __builtin_prefetch(Af + 32, 0, 2);   // global_prefetch_b8
      } else {
        const unsigned short* Ah =
            (const unsigned short*)Aptr + (size_t)(blockRow + rowA) * K + (k0 + kcA);
        uint4 v = *(const uint4*)Ah;
        dst[0] = v.x; dst[1] = v.y; dst[2] = v.z; dst[3] = v.w;
        if (k0 + 32 < K) __builtin_prefetch(Ah + 32, 0, 2);
      }
    }
    { // stage B tile = W rows (256x32), one full row per thread
      const unsigned short* Wp = W + (size_t)(blockCol + tid) * K + k0;
      unsigned* dst = (unsigned*)&sB[tid * 34];
      #pragma unroll
      for (int c = 0; c < 4; ++c) {
        uint4 w = ((const uint4*)Wp)[c];
        dst[c * 4 + 0] = w.x; dst[c * 4 + 1] = w.y;
        dst[c * 4 + 2] = w.z; dst[c * 4 + 3] = w.w;
      }
      if (k0 + 32 < K) __builtin_prefetch(Wp + 32, 0, 2);     // global_prefetch_b8
    }
    __syncthreads();

    Frag aF[2], bF[4];
    #pragma unroll
    for (int i = 0; i < 2; ++i) {                 // A frag: lane=row, paired K
      const int ar = waveM * 32 + i * 16 + l16;
      #pragma unroll
      for (int g = 0; g < 8; ++g) {
        const int kp = 2 * g + 16 * half;
        aF[i].u[g] = *(const unsigned*)&sA[ar * 34 + kp];
      }
    }
    #pragma unroll
    for (int j = 0; j < 4; ++j) {                 // B frag: lane=col, contig K
      const int br = waveN * 64 + j * 16 + l16;
      const unsigned short* src = &sB[br * 34 + 16 * half];
      #pragma unroll
      for (int e = 0; e < 8; ++e) bF[j].u[e] = *(const unsigned*)&src[2 * e];
    }
    #pragma unroll
    for (int i = 0; i < 2; ++i)
      #pragma unroll
      for (int j = 0; j < 4; ++j)
        acc[i][j] = wmma_bf16(aF[i].v, bF[j].v, acc[i][j]);
  }

  #pragma unroll
  for (int i = 0; i < 2; ++i)
    #pragma unroll
    for (int j = 0; j < 4; ++j) {
      const int col = blockCol + waveN * 64 + j * 16 + l16;
      const float bv = bias[col];
      #pragma unroll
      for (int r = 0; r < 8; ++r) {
        const int row = blockRow + waveM * 32 + i * 16 + r + 8 * half;
        float v = acc[i][j][r] + bv;
        if constexpr (RELU) v = fmaxf(v, 0.0f);
        if constexpr (OF32) ((float*)Out)[(size_t)row * N + col] = v;
        else ((unsigned short*)Out)[(size_t)row * N + col] = f2bf(v);
      }
    }
}

// ---------------------------------------------------------------------------
// Flash attention: one block = (b, head, 64-query tile); 4 waves, 16 q-rows each.
// qkv: bf16 [MTOK, 768] with q|k|v slices; Obuf: bf16 [MTOK, 256].
// ---------------------------------------------------------------------------
__global__ __launch_bounds__(128, 2)
void flash_attn(const unsigned short* __restrict__ qkv,
                unsigned short* __restrict__ Obuf) {
  __shared__ unsigned short sK[64 * 34];        // [key][dh], pad 34
  __shared__ unsigned short sVt[32 * 68];       // [dh][key], pad 68
  __shared__ unsigned short sP[4][16 * 68];     // per-wave P tile [16][64], pad 68

  const int tid = threadIdx.x, lane = tid & 31, wave = tid >> 5;
  const int half = lane >> 4, l16 = lane & 15;
  const int qt = blockIdx.x, head = blockIdx.y, b = blockIdx.z;
  const size_t tok0 = (size_t)b * SEQ;
  const int qcol = head * 32;

  Frag qF;  // Q fragment (16x32), lives in VGPRs for whole kernel
  {
    const size_t qr = (tok0 + qt * 64 + wave * 16 + l16) * TD + qcol;
    #pragma unroll
    for (int g = 0; g < 8; ++g) {
      const int kp = 2 * g + 16 * half;
      qF.u[g] = *(const unsigned*)&qkv[qr + kp];
    }
  }

  float mrun[8], lrun[8];
  #pragma unroll
  for (int r = 0; r < 8; ++r) { mrun[r] = -1e30f; lrun[r] = 0.0f; }
  const v8f vzero = {0.f,0.f,0.f,0.f,0.f,0.f,0.f,0.f};
  v8f oacc[2] = {vzero, vzero};
  const float scale = 0.17677669529663687f;     // 1/sqrt(32)

  for (int kb = 0; kb < SEQ / 64; ++kb) {
    __syncthreads();
    #pragma unroll
    for (int part = 0; part < 2; ++part) {      // stage K (row-major) + V (transposed)
      const int idx = part * 128 + tid;
      const int rowk = idx >> 2, chunk = (idx & 3) * 8;
      const size_t base = (tok0 + kb * 64 + rowk) * TD + qcol + chunk;
      union { uint4 q; unsigned w[4]; unsigned short s[8]; } kv;
      kv.q = *(const uint4*)&qkv[base + 256];   // K slice
      unsigned* kd = (unsigned*)&sK[rowk * 34 + chunk];
      kd[0] = kv.w[0]; kd[1] = kv.w[1]; kd[2] = kv.w[2]; kd[3] = kv.w[3];
      kv.q = *(const uint4*)&qkv[base + 512];   // V slice -> transposed
      #pragma unroll
      for (int e = 0; e < 8; ++e) sVt[(chunk + e) * 68 + rowk] = kv.s[e];
    }
    __syncthreads();

    v8f sc[4];                                  // scores 16x64 (4 col tiles)
    #pragma unroll
    for (int j = 0; j < 4; ++j) {
      Frag bF;
      const unsigned short* src = &sK[(j * 16 + l16) * 34 + 16 * half];
      #pragma unroll
      for (int e = 0; e < 8; ++e) bF.u[e] = *(const unsigned*)&src[2 * e];
      sc[j] = wmma_bf16(qF.v, bF.v, vzero);
    }
    #pragma unroll
    for (int j = 0; j < 4; ++j)
      #pragma unroll
      for (int r = 0; r < 8; ++r) sc[j][r] *= scale;

    float alpha[8], rsum[8];
    #pragma unroll
    for (int r = 0; r < 8; ++r) {               // online-softmax stats per row
      float m = fmaxf(fmaxf(sc[0][r], sc[1][r]), fmaxf(sc[2][r], sc[3][r]));
      #pragma unroll
      for (int off = 8; off; off >>= 1) m = fmaxf(m, __shfl_xor(m, off, 32));
      const float nm = fmaxf(mrun[r], m);
      alpha[r] = __expf(mrun[r] - nm);
      mrun[r] = nm;
      rsum[r] = 0.0f;
    }
    #pragma unroll
    for (int j = 0; j < 4; ++j)
      #pragma unroll
      for (int r = 0; r < 8; ++r) {
        const float p = __expf(sc[j][r] - mrun[r]);
        rsum[r] += p;
        sP[wave][(r + 8 * half) * 68 + j * 16 + l16] = f2bf(p);
      }
    #pragma unroll
    for (int r = 0; r < 8; ++r) {
      float s = rsum[r];
      #pragma unroll
      for (int off = 8; off; off >>= 1) s += __shfl_xor(s, off, 32);
      lrun[r] = lrun[r] * alpha[r] + s;
    }
    #pragma unroll
    for (int t = 0; t < 2; ++t)
      #pragma unroll
      for (int r = 0; r < 8; ++r) oacc[t][r] *= alpha[r];
    __syncthreads();                            // P tiles visible

    #pragma unroll
    for (int st = 0; st < 2; ++st) {            // O += P(16x64) @ V(64x32)
      Frag pF;
      #pragma unroll
      for (int g = 0; g < 8; ++g) {
        const int kp = 2 * g + 16 * half + 32 * st;
        pF.u[g] = *(const unsigned*)&sP[wave][l16 * 68 + kp];
      }
      #pragma unroll
      for (int t = 0; t < 2; ++t) {
        Frag vF;
        const unsigned short* src = &sVt[(t * 16 + l16) * 68 + st * 32 + 16 * half];
        #pragma unroll
        for (int e = 0; e < 8; ++e) vF.u[e] = *(const unsigned*)&src[2 * e];
        oacc[t] = wmma_bf16(pF.v, vF.v, oacc[t]);
      }
    }
  }

  #pragma unroll
  for (int t = 0; t < 2; ++t)
    #pragma unroll
    for (int r = 0; r < 8; ++r) {
      const int qr = qt * 64 + wave * 16 + r + 8 * half;
      const float v = oacc[t][r] / lrun[r];
      Obuf[(tok0 + qr) * DMODEL + qcol + t * 16 + l16] = f2bf(v);
    }
}

// ---------------------------------------------------------------------------
// h = LayerNorm(h + branch) * gamma + beta   (one row per wave, D=256)
// ---------------------------------------------------------------------------
__global__ __launch_bounds__(256)
void ln_residual(float* __restrict__ h, const float* __restrict__ br,
                 const float* __restrict__ gamma, const float* __restrict__ beta) {
  const int lane = threadIdx.x & 31;
  const int row  = blockIdx.x * 8 + (threadIdx.x >> 5);
  const size_t base = (size_t)row * DMODEL;
  float v[8], s = 0.f, sq = 0.f;
  #pragma unroll
  for (int k = 0; k < 8; ++k) {
    const int c = lane + 32 * k;
    v[k] = h[base + c] + br[base + c];
    s += v[k]; sq += v[k] * v[k];
  }
  #pragma unroll
  for (int off = 16; off; off >>= 1) {
    s  += __shfl_xor(s,  off, 32);
    sq += __shfl_xor(sq, off, 32);
  }
  const float mean = s * (1.0f / 256.0f);
  const float var  = sq * (1.0f / 256.0f) - mean * mean;
  const float rstd = rsqrtf(var + 1e-5f);
  #pragma unroll
  for (int k = 0; k < 8; ++k) {
    const int c = lane + 32 * k;
    h[base + c] = (v[k] - mean) * rstd * gamma[c] + beta[c];
  }
}

// ---------------------------------------------------------------------------
// Classifier head: out[MTOK,3] = h @ fc_w^T + fc_b   (tiny; plain fp32)
// ---------------------------------------------------------------------------
__global__ void fc_kernel(const float* __restrict__ h, const float* __restrict__ fw,
                          const float* __restrict__ fb, float* __restrict__ out) {
  const int idx = blockIdx.x * 256 + threadIdx.x;   // < MTOK*3
  const int row = idx / 3, c = idx - row * 3;
  const float* hr = h + (size_t)row * DMODEL;
  const float* wr = fw + c * DMODEL;
  float s = fb[c];
  #pragma unroll 4
  for (int k = 0; k < DMODEL; ++k) s += hr[k] * wr[k];
  out[idx] = s;
}

// ---------------------------------------------------------------------------
// Host orchestration
// ---------------------------------------------------------------------------
extern "C" void kernel_launch(void* const* d_in, const int* in_sizes, int n_in,
                              void* d_out, int out_size, void* d_ws, size_t ws_size,
                              hipStream_t stream) {
  (void)in_sizes; (void)n_in; (void)out_size; (void)ws_size;
  const int*   x          = (const int*)  d_in[0];
  const float* proj_w     = (const float*)d_in[1];
  const float* proj_b     = (const float*)d_in[2];
  const float* pos        = (const float*)d_in[3];
  const float* in_proj_w  = (const float*)d_in[4];
  const float* in_proj_b  = (const float*)d_in[5];
  const float* out_proj_w = (const float*)d_in[6];
  const float* out_proj_b = (const float*)d_in[7];
  const float* lin1_w     = (const float*)d_in[8];
  const float* lin1_b     = (const float*)d_in[9];
  const float* lin2_w     = (const float*)d_in[10];
  const float* lin2_b     = (const float*)d_in[11];
  const float* n1g        = (const float*)d_in[12];
  const float* n1b        = (const float*)d_in[13];
  const float* n2g        = (const float*)d_in[14];
  const float* n2b        = (const float*)d_in[15];
  const float* fc_w       = (const float*)d_in[16];
  const float* fc_b       = (const float*)d_in[17];

  char* ws = (char*)d_ws;
  float*          h    = (float*)         (ws + 0);           // 16 MB
  float*          tmp  = (float*)         (ws + 16777216);    // 16 MB
  unsigned short* qkv  = (unsigned short*)(ws + 33554432);    // 24 MB
  unsigned short* Ob   = (unsigned short*)(ws + 58720256);    //  8 MB
  unsigned short* hid  = (unsigned short*)(ws + 67108864);    // 64 MB
  unsigned short* wIn  = (unsigned short*)(ws + 134217728);
  unsigned short* wOut = (unsigned short*)(ws + 135004160);
  unsigned short* w1   = (unsigned short*)(ws + 135266304);
  unsigned short* w2   = (unsigned short*)(ws + 137363456);

  cvt_f32_bf16<<<(2*TD*DMODEL   + 255)/256, 256, 0, stream>>>(in_proj_w,  wIn,  2*TD*DMODEL);
  cvt_f32_bf16<<<(2*DMODEL*DMODEL+255)/256, 256, 0, stream>>>(out_proj_w, wOut, 2*DMODEL*DMODEL);
  cvt_f32_bf16<<<(2*FFD*DMODEL  + 255)/256, 256, 0, stream>>>(lin1_w,     w1,   2*FFD*DMODEL);
  cvt_f32_bf16<<<(2*DMODEL*FFD  + 255)/256, 256, 0, stream>>>(lin2_w,     w2,   2*DMODEL*FFD);

  embed_kernel<<<MTOK*DMODEL/256, 256, 0, stream>>>(x, proj_w, proj_b, pos, h);

  for (int l = 0; l < 2; ++l) {
    gemm_wmma<true,false,false><<<dim3(TD/256, MTOK/64), 256, 0, stream>>>(
        h, wIn + (size_t)l*TD*DMODEL, in_proj_b + l*TD, qkv, MTOK, TD, DMODEL);
    flash_attn<<<dim3(SEQ/64, 8, 16), 128, 0, stream>>>(qkv, Ob);
    gemm_wmma<false,true,false><<<dim3(DMODEL/256, MTOK/64), 256, 0, stream>>>(
        Ob, wOut + (size_t)l*DMODEL*DMODEL, out_proj_b + l*DMODEL, tmp, MTOK, DMODEL, DMODEL);
    ln_residual<<<MTOK/8, 256, 0, stream>>>(h, tmp, n1g + l*DMODEL, n1b + l*DMODEL);
    gemm_wmma<true,false,true><<<dim3(FFD/256, MTOK/64), 256, 0, stream>>>(
        h, w1 + (size_t)l*FFD*DMODEL, lin1_b + l*FFD, hid, MTOK, FFD, DMODEL);
    gemm_wmma<false,true,false><<<dim3(DMODEL/256, MTOK/64), 256, 0, stream>>>(
        hid, w2 + (size_t)l*DMODEL*FFD, lin2_b + l*DMODEL, tmp, MTOK, DMODEL, FFD);
    ln_residual<<<MTOK/8, 256, 0, stream>>>(h, tmp, n2g + l*DMODEL, n2b + l*DMODEL);
  }

  fc_kernel<<<MTOK*3/256, 256, 0, stream>>>(h, fc_w, fc_b, (float*)d_out);
}
